// MambaRPPGModel_44392781971765
// MI455X (gfx1250) — matrix operand order repaired and checked by reading, not compiled, not run on hardware
//
#include <hip/hip_runtime.h>
#include <hip/hip_bf16.h>

typedef __attribute__((ext_vector_type(16))) _Float16 v16h;
typedef __attribute__((ext_vector_type(8)))  float    v8f;
typedef __attribute__((ext_vector_type(4)))  float    v4f;
typedef __attribute__((ext_vector_type(4)))  unsigned int v4u;
typedef __attribute__((ext_vector_type(4)))  int      v4i;
typedef __attribute__((ext_vector_type(8)))  int      v8i;

#define DMODEL  160
#define DSTATE  64
#define DCONV   12
#define NBLOCKS 8
#define DINNER  320
#define DTRANK  10
#define KPRE    9
#define BB      8
#define LL      512
#define NTOK    (BB * LL)      // 4096 tokens
#define LDDBL   144            // padded row stride for dbl (138 -> 144)

// GEMM tiling
#define TM 32
#define TN 64
#define TK 32
#define LDS_STRIDE 33          // 32 + 1 pad DWORD per row (TDM pad_interval=32dw, pad=1dw)

// ---------------------------------------------------------------------------
// Math helpers
// ---------------------------------------------------------------------------
__device__ __forceinline__ float gelu_exact(float x) {
    return 0.5f * x * (1.0f + erff(x * 0.70710678118654752f));
}
__device__ __forceinline__ float silu(float x) {
    return x / (1.0f + __expf(-x));
}
__device__ __forceinline__ float softplus(float x) {
    return (x > 20.0f) ? x : log1pf(__expf(x));
}

// ---------------------------------------------------------------------------
// 1. Pre-conv (K=9, same padding) + bias + exact GELU : x(B,L,1) -> h(B,L,160)
// ---------------------------------------------------------------------------
__global__ __launch_bounds__(256) void pre_conv_gelu_kernel(
    const float* __restrict__ x, const float* __restrict__ pre_w,
    const float* __restrict__ pre_b, float* __restrict__ h)
{
    int idx = blockIdx.x * blockDim.x + threadIdx.x;       // over NTOK*DMODEL
    if (idx >= NTOK * DMODEL) return;
    int t = idx / DMODEL;
    int c = idx % DMODEL;
    int b = t / LL, l = t % LL;
    const float* xb = x + b * LL;
    float acc = pre_b[c];
    #pragma unroll
    for (int k = 0; k < KPRE; ++k) {
        int ll = l - (KPRE / 2) + k;
        if (ll >= 0 && ll < LL) acc += xb[ll] * pre_w[c * KPRE + k];
    }
    h[idx] = gelu_exact(acc);
}

// ---------------------------------------------------------------------------
// 2. LayerNorm over D_MODEL=160: one wave per token (160 = 5*32)
// ---------------------------------------------------------------------------
__global__ __launch_bounds__(256) void layernorm_kernel(
    const float* __restrict__ h, const float* __restrict__ w,
    const float* __restrict__ bia, float* __restrict__ xn)
{
    int lane = threadIdx.x & 31;
    int wave = threadIdx.x >> 5;
    int t = blockIdx.x * 8 + wave;
    if (t >= NTOK) return;
    const float* row = h + t * DMODEL;
    float v[5], s = 0.f, s2 = 0.f;
    #pragma unroll
    for (int j = 0; j < 5; ++j) {
        v[j] = row[lane + 32 * j];
        s += v[j];
        s2 += v[j] * v[j];
    }
    #pragma unroll
    for (int off = 16; off > 0; off >>= 1) {
        s  += __shfl_xor(s, off, 32);
        s2 += __shfl_xor(s2, off, 32);
    }
    float mu  = s * (1.0f / DMODEL);
    float var = s2 * (1.0f / DMODEL) - mu * mu;
    float rstd = rsqrtf(var + 1e-5f);
    float* orow = xn + t * DMODEL;
    #pragma unroll
    for (int j = 0; j < 5; ++j) {
        int c = lane + 32 * j;
        orow[c] = (v[j] - mu) * rstd * w[c] + bia[c];
    }
}

// ---------------------------------------------------------------------------
// TDM tile load: rect tile [rows_tile x TK] of row-major f32 matrix (row
// stride ld elems) -> LDS at ldsOff, LDS row stride = 33 DWORDs (pad 1 DWORD
// per 32, set in D#). Rows >= rows_valid hardware-zero-filled (tensor_dim1).
// ---------------------------------------------------------------------------
#if __has_builtin(__builtin_amdgcn_tensor_load_to_lds)
#define USE_TDM 1
__device__ __forceinline__ void tdm_load_tile_f32(
    const float* gptr, unsigned ldsOff, int rows_valid, int rows_tile, int ld)
{
    unsigned long long ga = (unsigned long long)(uintptr_t)gptr;
    v4u g0;
    g0.x = 1u;                                         // count=1, user descriptor
    g0.y = ldsOff;                                     // lds_addr (bytes)
    g0.z = (unsigned)(ga & 0xFFFFFFFFull);             // global_addr[31:0]
    g0.w = (unsigned)((ga >> 32) & 0x1FFFFFFull)       // global_addr[56:32]
         | (2u << 30);                                 // type=2 ("image")
    v8i g1;
    g1[0] = (int)((2u << 16)      // data_size = 4B
                | (1u << 20)      // pad_enable
                | (4u << 22)      // pad_interval: every 32 DWORDs
                | (0u << 25));    // pad_amount: 1 DWORD
    unsigned td0 = (unsigned)TK;               // tensor_dim0 (32-bit)
    unsigned td1 = (unsigned)rows_valid;       // tensor_dim1 (32-bit)
    g1[1] = (int)((td0 & 0xFFFFu) << 16);                          // [63:48]=td0.lo
    g1[2] = (int)(((td0 >> 16) & 0xFFFFu) | ((td1 & 0xFFFFu) << 16));
    g1[3] = (int)(((td1 >> 16) & 0xFFFFu) | ((unsigned)TK << 16)); // tile_dim0=32
    g1[4] = (int)((unsigned)rows_tile & 0xFFFFu);                  // tile_dim1; tile_dim2=0
    g1[5] = (int)(unsigned)ld;                                     // dim0_stride[31:0]
    g1[6] = 0;
    g1[7] = 0;
    v4i z4 = {0, 0, 0, 0};
#if defined(__clang_major__) && (__clang_major__ >= 23)
    v8i z8 = {0, 0, 0, 0, 0, 0, 0, 0};
    __builtin_amdgcn_tensor_load_to_lds(g0, g1, z4, z4, z8, 0);
#else
    __builtin_amdgcn_tensor_load_to_lds(g0, g1, z4, z4, 0);
#endif
}
#else
#define USE_TDM 0
#endif

// ---------------------------------------------------------------------------
// 3. WMMA GEMM: C[M,N] = A[M,K] * B[N,K]^T (+ optional residual)
//    Double-buffered TDM pipeline: wave 0 issues next k-step's tile DMA, then
//    waits tensorcnt<=2 (current buffer's pair complete, in-order), barrier
//    publishes LDS to all 8 waves; fragments cvt f32->f16 in regs; f32
//    accumulate via v_wmma_f32_16x16x32_f16. Wave (wm,wn) owns a 16x16 tile
//    of the 32(M) x 64(N) workgroup tile. Requires M%32==0, K%32==0.
// ---------------------------------------------------------------------------
__global__ __launch_bounds__(256) void gemm_wmma_kernel(
    const float* __restrict__ A, const float* __restrict__ Bw,
    const float* __restrict__ Cres, float* __restrict__ C,
    int M, int N, int K, int lda, int ldb, int ldc)
{
    __shared__ float AsF[2][TM * LDS_STRIDE];   // 2 x 32x32, row stride 33
    __shared__ float BsF[2][TN * LDS_STRIDE];   // 2 x 64x32, row stride 33

    const int tid  = threadIdx.x;
    const int lane = tid & 31;
    const int wave = tid >> 5;
    const int wm   = wave & 1;
    const int wn   = wave >> 1;
    const int m0   = blockIdx.y * TM;
    const int n0   = blockIdx.x * TN;
    const int hi   = (lane >> 4) & 1;
    const int l15  = lane & 15;
    const int rowsB = (N - n0 < TN) ? (N - n0) : TN;   // valid B rows this block
    const int ksteps = K / TK;

    v8f acc = {};

#if USE_TDM
    if (wave == 0) {   // prologue: k-step 0 into buffer 0
        tdm_load_tile_f32(A  + (size_t)m0 * lda,
                          (unsigned)(uintptr_t)(void*)AsF[0], TM, TM, lda);
        tdm_load_tile_f32(Bw + (size_t)n0 * ldb,
                          (unsigned)(uintptr_t)(void*)BsF[0], rowsB, TN, ldb);
    }
#endif

    for (int ks = 0; ks < ksteps; ++ks) {
        const int cur = ks & 1;
#if USE_TDM
        if (wave == 0) {
            if (ks + 1 < ksteps) {
                int k1 = (ks + 1) * TK;
                tdm_load_tile_f32(A  + (size_t)m0 * lda + k1,
                                  (unsigned)(uintptr_t)(void*)AsF[cur ^ 1], TM, TM, lda);
                tdm_load_tile_f32(Bw + (size_t)n0 * ldb + k1,
                                  (unsigned)(uintptr_t)(void*)BsF[cur ^ 1], rowsB, TN, ldb);
                // 2 newest ops may remain in flight; current pair done (in-order)
                __builtin_amdgcn_s_wait_tensorcnt(2);
            } else {
                __builtin_amdgcn_s_wait_tensorcnt(0);
            }
        }
        __syncthreads();   // publish DMA'd tiles to all waves
#else
        // Fallback: branch-free vectorized staging (float4 -> LDS f32)
        {
            int k0 = ks * TK;
            int r = tid >> 3, q = tid & 7;             // A: 32 rows x 8 qwords
            v4f va = *(const v4f*)(A + (size_t)(m0 + r) * lda + k0 + q * 4);
            *(v4f*)&AsF[0][r * LDS_STRIDE + q * 4] = va;
            #pragma unroll
            for (int it = 0; it < 2; ++it) {           // B: 64 rows x 8 qwords
                int t2 = tid + it * 256;
                int rb = t2 >> 3, qb = t2 & 7;
                int gn = n0 + rb;
                int gcl = (gn < N) ? gn : 0;
                float msk = (gn < N) ? 1.0f : 0.0f;
                v4f vb = *(const v4f*)(Bw + (size_t)gcl * ldb + k0 + qb * 4);
                vb *= msk;
                *(v4f*)&BsF[0][rb * LDS_STRIDE + qb * 4] = vb;
            }
        }
        __syncthreads();
#endif

        // Fragments per ISA 7.12.2.
        // A (16x32): lane<16 row=l15 K in {0..7,16..23}; lane>=16 K in {8..15,24..31}
        v16h af, bf;
        const float* arow = &AsF[cur][(wm * 16 + l15) * LDS_STRIDE];
        #pragma unroll
        for (int j = 0; j < 16; ++j) {
            int kk = j + ((j >= 8) ? 8 : 0) + (hi ? 8 : 0);
            af[j] = (_Float16)arow[kk];
        }
        // B (32x16): lane<16 col=l15 K in {0..7,16..23}; lane>=16 K in {8..15,24..31}
        const float* brow = &BsF[cur][(wn * 16 + l15) * LDS_STRIDE];
        #pragma unroll
        for (int j = 0; j < 16; ++j) {
            int kk = j + ((j >= 8) ? 8 : 0) + (hi ? 8 : 0);
            bf[j] = (_Float16)brow[kk];
        }

        acc = __builtin_amdgcn_wmma_f32_16x16x32_f16(
            false, af, false, bf, (short)0, acc, false, false);

        __syncthreads();   // all waves done reading buffer `cur` before rewrite
    }

    // C layout: lane<16 -> N=l15, VGPR r -> M=r ; lane>=16 -> M=8+r
    int ncol  = n0 + wn * 16 + l15;
    int mbase = m0 + wm * 16 + (hi ? 8 : 0);
    if (ncol < N) {
        if (Cres) {
            float res[8];
            #pragma unroll
            for (int r = 0; r < 8; ++r)     // batched load clause, single wait
                res[r] = Cres[(size_t)(mbase + r) * ldc + ncol];
            #pragma unroll
            for (int r = 0; r < 8; ++r)
                C[(size_t)(mbase + r) * ldc + ncol] = acc[r] + res[r];
        } else {
            #pragma unroll
            for (int r = 0; r < 8; ++r)
                C[(size_t)(mbase + r) * ldc + ncol] = acc[r];
        }
    }
}

// ---------------------------------------------------------------------------
// 4. Causal depthwise conv (K=12) + bias + SiLU on u = xz[:, :, :320]
// ---------------------------------------------------------------------------
__global__ __launch_bounds__(256) void conv_silu_kernel(
    const float* __restrict__ xz, const float* __restrict__ cw,
    const float* __restrict__ cb, float* __restrict__ u)
{
    int idx = blockIdx.x * blockDim.x + threadIdx.x;
    if (idx >= NTOK * DINNER) return;
    int t = idx / DINNER, d = idx % DINNER;
    int b = t / LL, l = t % LL;
    const float* base = xz + (size_t)b * LL * (2 * DINNER) + d;
    float acc = cb[d];
    #pragma unroll
    for (int k = 0; k < DCONV; ++k) {
        int ll = l - (DCONV - 1) + k;
        if (ll >= 0) acc += base[(size_t)ll * (2 * DINNER)] * cw[d * DCONV + k];
    }
    u[idx] = silu(acc);
}

// ---------------------------------------------------------------------------
// 5. dt = softplus(dtr @ dt_proj_w^T + dt_proj_b)  (K = DT_RANK = 10)
// ---------------------------------------------------------------------------
__global__ __launch_bounds__(256) void dt_kernel(
    const float* __restrict__ dbl, const float* __restrict__ dtw,
    const float* __restrict__ dtb, float* __restrict__ dt)
{
    int idx = blockIdx.x * blockDim.x + threadIdx.x;
    if (idx >= NTOK * DINNER) return;
    int t = idx / DINNER, d = idx % DINNER;
    const float* row = dbl + t * LDDBL;
    float acc = dtb[d];
    #pragma unroll
    for (int r = 0; r < DTRANK; ++r) acc += row[r] * dtw[d * DTRANK + r];
    dt[idx] = softplus(acc);
}

// ---------------------------------------------------------------------------
// 6. Selective scan fused with D-skip and silu(z) gating.
//    One wave per (b, d); lane holds states n=lane and n=lane+32.
//    Software-pipelined: next step's operands load during current reduce.
// ---------------------------------------------------------------------------
__global__ __launch_bounds__(256) void scan_kernel(
    const float* __restrict__ u, const float* __restrict__ dt,
    const float* __restrict__ dbl, const float* __restrict__ A_log,
    const float* __restrict__ D_skip, const float* __restrict__ xz,
    float* __restrict__ y)
{
    const int lane = threadIdx.x & 31;
    const int wave = threadIdx.x >> 5;
    const int b    = blockIdx.x / (DINNER / 8);
    const int dch  = blockIdx.x % (DINNER / 8);
    const int d    = dch * 8 + wave;

    const float A0 = -__expf(A_log[d * DSTATE + lane]);
    const float A1 = -__expf(A_log[d * DSTATE + lane + 32]);
    const float Dv = D_skip[d];

    const float* ub   = u   + (size_t)b * LL * DINNER + d;
    const float* dtb_ = dt  + (size_t)b * LL * DINNER + d;
    const float* dblb = dbl + (size_t)b * LL * LDDBL + DTRANK;
    const float* zb   = xz  + (size_t)b * LL * (2 * DINNER) + DINNER + d;
    float*       yb   = y   + (size_t)b * LL * DINNER + d;

    // prologue loads (l = 0)
    float dtv = dtb_[0];
    float uv  = ub[0];
    float Bv0 = dblb[lane];
    float Bv1 = dblb[lane + 32];
    float Cv0 = dblb[DSTATE + lane];
    float Cv1 = dblb[DSTATE + lane + 32];
    float zv  = zb[0];

    float h0 = 0.f, h1 = 0.f;
    for (int l = 0; l < LL; ++l) {
        // issue next step's loads early to overlap the serial exp/reduce chain
        float ndt = 0.f, nuv = 0.f, nB0 = 0.f, nB1 = 0.f, nC0 = 0.f, nC1 = 0.f, nz = 0.f;
        if (l + 1 < LL) {
            ndt = dtb_[(size_t)(l + 1) * DINNER];
            nuv = ub[(size_t)(l + 1) * DINNER];
            const float* nrow = dblb + (size_t)(l + 1) * LDDBL;
            nB0 = nrow[lane];          nB1 = nrow[lane + 32];
            nC0 = nrow[DSTATE + lane]; nC1 = nrow[DSTATE + lane + 32];
            nz  = zb[(size_t)(l + 1) * (2 * DINNER)];
        }
        float du = dtv * uv;
        h0 = __expf(dtv * A0) * h0 + du * Bv0;
        h1 = __expf(dtv * A1) * h1 + du * Bv1;
        float p = h0 * Cv0 + h1 * Cv1;
        #pragma unroll
        for (int off = 16; off > 0; off >>= 1) p += __shfl_xor(p, off, 32);
        if (lane == 0) yb[(size_t)l * DINNER] = (p + uv * Dv) * silu(zv);

        dtv = ndt; uv = nuv; Bv0 = nB0; Bv1 = nB1; Cv0 = nC0; Cv1 = nC1; zv = nz;
    }
}

// ---------------------------------------------------------------------------
// 7. Head: out[t] = h[t,:] . head_w + head_b   (one wave per token)
// ---------------------------------------------------------------------------
__global__ __launch_bounds__(256) void head_kernel(
    const float* __restrict__ h, const float* __restrict__ hw,
    const float* __restrict__ hb, float* __restrict__ out)
{
    int lane = threadIdx.x & 31;
    int wave = threadIdx.x >> 5;
    int t = blockIdx.x * 8 + wave;
    if (t >= NTOK) return;
    const float* row = h + t * DMODEL;
    float s = 0.f;
    #pragma unroll
    for (int j = 0; j < 5; ++j) s += row[lane + 32 * j] * hw[lane + 32 * j];
    #pragma unroll
    for (int off = 16; off > 0; off >>= 1) s += __shfl_xor(s, off, 32);
    if (lane == 0) out[t] = s + hb[0];
}

// ---------------------------------------------------------------------------
// Host-side launch
// ---------------------------------------------------------------------------
extern "C" void kernel_launch(void* const* d_in, const int* in_sizes, int n_in,
                              void* d_out, int out_size, void* d_ws, size_t ws_size,
                              hipStream_t stream) {
    const float* x        = (const float*)d_in[0];
    const float* pre_w    = (const float*)d_in[1];
    const float* pre_b    = (const float*)d_in[2];
    const float* ln_w     = (const float*)d_in[3];
    const float* ln_b     = (const float*)d_in[4];
    const float* in_projw = (const float*)d_in[5];
    const float* conv_w   = (const float*)d_in[6];
    const float* conv_b   = (const float*)d_in[7];
    const float* x_projw  = (const float*)d_in[8];
    const float* dt_projw = (const float*)d_in[9];
    const float* dt_projb = (const float*)d_in[10];
    const float* A_log    = (const float*)d_in[11];
    const float* D_skip   = (const float*)d_in[12];
    const float* out_projw= (const float*)d_in[13];
    const float* head_w   = (const float*)d_in[14];
    const float* head_b   = (const float*)d_in[15];
    float* out = (float*)d_out;

    float* ws  = (float*)d_ws;
    float* h   = ws;                                  // 4096*160
    float* xn  = h   + (size_t)NTOK * DMODEL;         // 4096*160
    float* xz  = xn  + (size_t)NTOK * DMODEL;         // 4096*640
    float* u   = xz  + (size_t)NTOK * 2 * DINNER;     // 4096*320
    float* dbl = u   + (size_t)NTOK * DINNER;         // 4096*144
    float* dt  = dbl + (size_t)NTOK * LDDBL;          // 4096*320
    float* y   = dt  + (size_t)NTOK * DINNER;         // 4096*320

    pre_conv_gelu_kernel<<<(NTOK * DMODEL + 255) / 256, 256, 0, stream>>>(x, pre_w, pre_b, h);

    for (int i = 0; i < NBLOCKS; ++i) {
        layernorm_kernel<<<NTOK / 8, 256, 0, stream>>>(h, ln_w + i * DMODEL, ln_b + i * DMODEL, xn);

        // in_proj: (4096x160) @ (640x160)^T -> xz
        gemm_wmma_kernel<<<dim3((2 * DINNER) / TN, NTOK / TM), 256, 0, stream>>>(
            xn, in_projw + (size_t)i * 2 * DINNER * DMODEL, nullptr, xz,
            NTOK, 2 * DINNER, DMODEL, DMODEL, DMODEL, 2 * DINNER);

        conv_silu_kernel<<<(NTOK * DINNER + 255) / 256, 256, 0, stream>>>(
            xz, conv_w + (size_t)i * DINNER * DCONV, conv_b + (size_t)i * DINNER, u);

        // x_proj: (4096x320) @ (138x320)^T -> dbl (row stride 144)
        gemm_wmma_kernel<<<dim3((138 + TN - 1) / TN, NTOK / TM), 256, 0, stream>>>(
            u, x_projw + (size_t)i * 138 * DINNER, nullptr, dbl,
            NTOK, 138, DINNER, DINNER, DINNER, LDDBL);

        dt_kernel<<<(NTOK * DINNER + 255) / 256, 256, 0, stream>>>(
            dbl, dt_projw + (size_t)i * DINNER * DTRANK, dt_projb + (size_t)i * DINNER, dt);

        scan_kernel<<<BB * (DINNER / 8), 256, 0, stream>>>(
            u, dt, dbl, A_log + (size_t)i * DINNER * DSTATE, D_skip + (size_t)i * DINNER, xz, y);

        // out_proj + residual: h = h + y @ W^T
        gemm_wmma_kernel<<<dim3((DMODEL + TN - 1) / TN, NTOK / TM), 256, 0, stream>>>(
            y, out_projw + (size_t)i * DMODEL * DINNER, h, h,
            NTOK, DMODEL, DINNER, DINNER, DINNER, DMODEL);
    }

    head_kernel<<<NTOK / 8, 256, 0, stream>>>(h, head_w, head_b, out);
}
